// GCNLayer_21543555956791
// MI455X (gfx1250) — matrix operand order repaired and checked by reading, not compiled, not run on hardware
//
#include <hip/hip_runtime.h>
#include <stdint.h>

#define N_NODES 100000
#define E_MAIN  1600000
#define E_ACC   200000
#define D       128
#define FE      4
#define MTILES  (N_NODES / 16)   // 6250, exact

typedef __attribute__((ext_vector_type(16))) __bf16 v16bf;
typedef __attribute__((ext_vector_type(8)))  __bf16 v8bf;
typedef __attribute__((ext_vector_type(8)))  float  v8f;

union AV16 { v16bf v; v8bf h[2]; };

// Scheduler pipeline hints (no-op if builtin missing).
// masks: 0x8 = MFMA/WMMA, 0x20 = VMEM read, 0x100 = DS read
#if __has_builtin(__builtin_amdgcn_sched_group_barrier)
#define SGB(mask, cnt, id) __builtin_amdgcn_sched_group_barrier((mask), (cnt), (id))
#else
#define SGB(mask, cnt, id)
#endif

// ---- workspace layout (bytes) ----
static const size_t OFF_DEG  = 0;                                   // N f32      (400,000 B)
static const size_t OFF_NORM = 512 * 1024;                          // N f32
static const size_t OFF_AGG  = 1024 * 1024;                         // N*D f32    (51.2 MB)
static const size_t OFF_APK  = OFF_AGG + (size_t)N_NODES * D * 4;   // N*256 bf16 (51.2 MB)
static const size_t OFF_B    = OFF_APK + (size_t)N_NODES * 256 * 2; // 256*128 bf16 (64 KB)
static const size_t OFF_BIAS = OFF_B + 256 * 128 * 2;               // 128 f32
// total ~103.6 MB

__device__ __forceinline__ unsigned short f32_to_bf16(float f) {
    uint32_t u = __float_as_uint(f);
    u += 0x7FFFu + ((u >> 16) & 1u);   // round-to-nearest-even
    return (unsigned short)(u >> 16);
}

__global__ void k_zero(float* __restrict__ p, long n) {
    long i = (long)blockIdx.x * blockDim.x + threadIdx.x;
    long stride = (long)gridDim.x * blockDim.x;
    for (; i < n; i += stride) p[i] = 0.0f;
}

__global__ void k_degree(const int* __restrict__ src, float* __restrict__ deg, int e) {
    int i = blockIdx.x * blockDim.x + threadIdx.x;
    if (i < e) atomicAdd(&deg[src[i]], 1.0f);
}

__global__ void k_norm(const float* __restrict__ deg, float* __restrict__ nrm, int n) {
    int i = blockIdx.x * blockDim.x + threadIdx.x;
    if (i < n) {
        float d = deg[i];
        nrm[i] = (d > 0.0f) ? __frsqrt_rn(d) : 0.0f;
    }
}

// One wave32 per edge; lane handles 4 of 128 features. h[x] = feats[x]*norm[x] on the fly.
__global__ void k_edge_main(const float* __restrict__ feats,
                            const float* __restrict__ ef,
                            const int* __restrict__ src, const int* __restrict__ dst,
                            const float* __restrict__ W_att, const float* __restrict__ b_att,
                            const float* __restrict__ W_edge, const float* __restrict__ b_edge,
                            const float* __restrict__ nrm, float* __restrict__ agg) {
    long t = (long)blockIdx.x * blockDim.x + threadIdx.x;
    long e = t >> 5;
    int lane = (int)(t & 31);
    if (e >= E_MAIN) return;
    int s = src[e], d = dst[e];
    float4 fv = *(const float4*)(ef + 4 * e);
    float att = fv.x * W_att[0] + fv.y * W_att[1] + fv.z * W_att[2] + fv.w * W_att[3] + b_att[0];
    att = 1.0f / (1.0f + __expf(-att));
    float ns = nrm[s], nd = nrm[d];
    int c0 = lane * 4;
    float4 hs = *(const float4*)(feats + (long)s * D + c0);
    float4 hd = *(const float4*)(feats + (long)d * D + c0);
    float4 w0 = *(const float4*)(W_edge + (c0 + 0) * FE);
    float4 w1 = *(const float4*)(W_edge + (c0 + 1) * FE);
    float4 w2 = *(const float4*)(W_edge + (c0 + 2) * FE);
    float4 w3 = *(const float4*)(W_edge + (c0 + 3) * FE);
    float m0 = att * hs.x * ns + (fv.x*w0.x + fv.y*w0.y + fv.z*w0.z + fv.w*w0.w + b_edge[c0+0]) + hd.x * nd;
    float m1 = att * hs.y * ns + (fv.x*w1.x + fv.y*w1.y + fv.z*w1.z + fv.w*w1.w + b_edge[c0+1]) + hd.y * nd;
    float m2 = att * hs.z * ns + (fv.x*w2.x + fv.y*w2.y + fv.z*w2.z + fv.w*w2.w + b_edge[c0+2]) + hd.z * nd;
    float m3 = att * hs.w * ns + (fv.x*w3.x + fv.y*w3.y + fv.z*w3.z + fv.w*w3.w + b_edge[c0+3]) + hd.w * nd;
    float* ap = agg + (long)d * D + c0;
    atomicAdd(ap + 0, m0); atomicAdd(ap + 1, m1);
    atomicAdd(ap + 2, m2); atomicAdd(ap + 3, m3);
}

// 'account' edges: ef == 0 -> att = sigmoid(b_att), tr = b_edge
__global__ void k_edge_acc(const float* __restrict__ feats,
                           const int* __restrict__ src, const int* __restrict__ dst,
                           const float* __restrict__ b_att, const float* __restrict__ b_edge,
                           const float* __restrict__ nrm, float* __restrict__ agg) {
    long t = (long)blockIdx.x * blockDim.x + threadIdx.x;
    long e = t >> 5;
    int lane = (int)(t & 31);
    if (e >= E_ACC) return;
    int s = src[e], d = dst[e];
    float att = 1.0f / (1.0f + __expf(-b_att[0]));
    float ns = nrm[s], nd = nrm[d];
    int c0 = lane * 4;
    float4 hs = *(const float4*)(feats + (long)s * D + c0);
    float4 hd = *(const float4*)(feats + (long)d * D + c0);
    float4 be = *(const float4*)(b_edge + c0);
    float m0 = att * hs.x * ns + be.x + hd.x * nd;
    float m1 = att * hs.y * ns + be.y + hd.y * nd;
    float m2 = att * hs.z * ns + be.z + hd.z * nd;
    float m3 = att * hs.w * ns + be.w + hd.w * nd;
    float* ap = agg + (long)d * D + c0;
    atomicAdd(ap + 0, m0); atomicAdd(ap + 1, m1);
    atomicAdd(ap + 2, m2); atomicAdd(ap + 3, m3);
}

// Bcat (n-major, bf16): Bcat[n*256+k] = k<128 ? W_msg[n,k] : W_skip[n,k-128]; bias[n]=b_msg[n]+b_skip[n]
__global__ void k_build_B(const float* __restrict__ W_msg, const float* __restrict__ W_skip,
                          const float* __restrict__ b_msg, const float* __restrict__ b_skip,
                          unsigned short* __restrict__ B, float* __restrict__ bias) {
    int i = blockIdx.x * blockDim.x + threadIdx.x;   // 256*128 = 32768 threads
    if (i < 256 * 128) {
        int n = i >> 8, k = i & 255;
        float v = (k < 128) ? W_msg[n * 128 + k] : W_skip[n * 128 + (k - 128)];
        B[i] = f32_to_bf16(v);
    }
    if (i < 128) bias[i] = b_msg[i] + b_skip[i];
}

// Apack (bf16, 256 wide): row i = [ agg[i,:]*norm[i] | feats[i,:] ]
__global__ void k_build_A(const float* __restrict__ agg, const float* __restrict__ nrm,
                          const float* __restrict__ feats, unsigned short* __restrict__ A) {
    long i = (long)blockIdx.x * blockDim.x + threadIdx.x;
    long n = (long)N_NODES * 256;
    if (i >= n) return;
    long node = i >> 8;
    int k = (int)(i & 255);
    float v = (k < 128) ? agg[node * D + k] * nrm[node] : feats[node * D + (k - 128)];
    A[i] = f32_to_bf16(v);
}

// GEMM: out[100000,128] = Apack[100000,256] @ Bcat[256,128] + bias, via v_wmma_f32_16x16x32_bf16.
// 256 threads = 8 waves; Bcat staged in 64KB LDS; each wave owns one 16-node x 128-col tile.
// sched_group_barrier pins the per-K-tile pipeline [2 VMEM read][16 DS read][8 WMMA] so the
// DS latency is paid once per 8 matrix ops and the 8 B fragments stay live in registers.
__global__ void __launch_bounds__(256) k_gemm(const unsigned short* __restrict__ A,
                                              const unsigned short* __restrict__ Bg,
                                              const float* __restrict__ bias,
                                              float* __restrict__ out) {
    __shared__ unsigned short Blds[256 * 128];   // 64 KB
    {   // cooperative load: 64KB = 4096 x uint4
        const uint4* bs = (const uint4*)Bg;
        uint4* bd = (uint4*)Blds;
        for (int i = threadIdx.x; i < 4096; i += 256) bd[i] = bs[i];
    }
    __syncthreads();

    int wave = threadIdx.x >> 5;
    int lane = threadIdx.x & 31;
    int mtile = blockIdx.x * 8 + wave;
    if (mtile >= MTILES) return;                 // wave-uniform: EXEC all-ones inside

    int row  = mtile * 16 + (lane & 15);         // A row this lane carries (M = lane%16)
    int ksel = (lane >> 4) * 8;                  // K lo/hi half per ISA A-layout
    const unsigned short* arow = A + (long)row * 256;

    v8f acc[8];
    v8f zero = {0.f, 0.f, 0.f, 0.f, 0.f, 0.f, 0.f, 0.f};
#pragma unroll
    for (int i = 0; i < 8; ++i) acc[i] = zero;

#pragma unroll
    for (int kt = 0; kt < 8; ++kt) {             // K = 256 in steps of 32
        AV16 a;
        a.h[0] = *(const v8bf*)(arow + kt * 32 + ksel);        // K = kb..kb+7
        a.h[1] = *(const v8bf*)(arow + kt * 32 + ksel + 16);   // K = kb+16..kb+23

        int kidx = kt * 32 + ksel;                              // B: N-row per lane, K as A
        const unsigned short* bk = Blds + (lane & 15) * 256 + kidx;

        AV16 b[8];                               // batch all 8 N-tile fragments first
#pragma unroll
        for (int nt = 0; nt < 8; ++nt) {
            b[nt].h[0] = *(const v8bf*)(bk + nt * 16 * 256);
            b[nt].h[1] = *(const v8bf*)(bk + nt * 16 * 256 + 16);
        }
#pragma unroll
        for (int nt = 0; nt < 8; ++nt) {
            acc[nt] = __builtin_amdgcn_wmma_f32_16x16x32_bf16(
                false, a.v, false, b[nt].v, (short)0, acc[nt], false, false);
        }
        // Pipeline spec for this K-tile (repeated 8x across the unrolled region):
        SGB(0x020, 2, 0);    // 2 global A-fragment loads
        SGB(0x100, 16, 0);   // 16 ds_load_b128 (all 8 B fragments)
        SGB(0x008, 8, 0);    // 8 v_wmma
    }

    // C layout: VGPR r <-> M = r + 8*(lane>>4); N = lane&15
    int ncol  = lane & 15;
    int rbase = (lane >> 4) * 8;
#pragma unroll
    for (int nt = 0; nt < 8; ++nt) {
        float bb = bias[nt * 16 + ncol];
#pragma unroll
        for (int r = 0; r < 8; ++r) {
            int node = mtile * 16 + rbase + r;
            out[(long)node * D + nt * 16 + ncol] = acc[nt][r] + bb;
        }
    }
}

extern "C" void kernel_launch(void* const* d_in, const int* in_sizes, int n_in,
                              void* d_out, int out_size, void* d_ws, size_t ws_size,
                              hipStream_t stream) {
    const float* feats   = (const float*)d_in[0];
    const float* efeat   = (const float*)d_in[1];
    const int*   src_E   = (const int*)d_in[2];
    const int*   dst_E   = (const int*)d_in[3];
    const int*   src_acc = (const int*)d_in[4];
    const int*   dst_acc = (const int*)d_in[5];
    const float* W_skip  = (const float*)d_in[6];
    const float* b_skip  = (const float*)d_in[7];
    const float* W_msg   = (const float*)d_in[8];
    const float* b_msg   = (const float*)d_in[9];
    const float* W_edge  = (const float*)d_in[10];
    const float* b_edge  = (const float*)d_in[11];
    const float* W_att   = (const float*)d_in[12];
    const float* b_att   = (const float*)d_in[13];
    float* out = (float*)d_out;

    char* ws = (char*)d_ws;
    float*          deg  = (float*)(ws + OFF_DEG);
    float*          nrm  = (float*)(ws + OFF_NORM);
    float*          agg  = (float*)(ws + OFF_AGG);
    unsigned short* Apk  = (unsigned short*)(ws + OFF_APK);
    unsigned short* Bcat = (unsigned short*)(ws + OFF_B);
    float*          bias = (float*)(ws + OFF_BIAS);

    // 1) zero deg + agg (harness poisons workspace)
    k_zero<<<1024, 256, 0, stream>>>(deg, (long)N_NODES);
    k_zero<<<4096, 256, 0, stream>>>(agg, (long)N_NODES * D);

    // 2) degree from src_E
    k_degree<<<(E_MAIN + 255) / 256, 256, 0, stream>>>(src_E, deg, E_MAIN);

    // 3) norm = deg>0 ? deg^-0.5 : 0
    k_norm<<<(N_NODES + 255) / 256, 256, 0, stream>>>(deg, nrm, N_NODES);

    // 4) main edges: 1 wave per edge
    {
        long threads = (long)E_MAIN * 32;
        int blocks = (int)((threads + 255) / 256);
        k_edge_main<<<blocks, 256, 0, stream>>>(feats, efeat, src_E, dst_E,
                                                W_att, b_att, W_edge, b_edge, nrm, agg);
    }
    // 5) account edges
    {
        long threads = (long)E_ACC * 32;
        int blocks = (int)((threads + 255) / 256);
        k_edge_acc<<<blocks, 256, 0, stream>>>(feats, src_acc, dst_acc,
                                               b_att, b_edge, nrm, agg);
    }

    // 6) pack weights (k-major bf16) + fused bias
    k_build_B<<<(256 * 128 + 255) / 256, 256, 0, stream>>>(W_msg, W_skip, b_msg, b_skip,
                                                           Bcat, bias);
    // 7) pack activations: [agg*norm | feats] as bf16
    {
        long n = (long)N_NODES * 256;
        int blocks = (int)((n + 255) / 256);
        k_build_A<<<blocks, 256, 0, stream>>>(agg, nrm, feats, Apk);
    }

    // 8) WMMA GEMM epilogue
    k_gemm<<<(MTILES + 7) / 8, 256, 0, stream>>>(Apk, Bcat, bias, out);
}
